// GPT2Attention_40209483825263
// MI455X (gfx1250) — matrix operand order repaired and checked by reading, not compiled, not run on hardware
//
#include <hip/hip_runtime.h>
#include <hip/hip_bf16.h>

// ---- CDNA5 wave32 WMMA types ----
typedef __attribute__((ext_vector_type(16))) _Float16 v16h;
typedef __attribute__((ext_vector_type(8)))  _Float16 v8h;
typedef __attribute__((ext_vector_type(8)))  float    v8f;
// ---- TDM descriptor vector types (6-arg toolchain variant) ----
typedef unsigned int u32x4 __attribute__((ext_vector_type(4)));
typedef int          i32x8 __attribute__((ext_vector_type(8)));
typedef int          i32x4 __attribute__((ext_vector_type(4)));

#define DMODEL 768
#define NHEAD  12
#define HDIM   64
#define SEQT   4096
#define BATCH  2

// ---------------------------------------------------------------------------
// Fragment loaders matching CDNA5 ISA VGPR layouts (CDNA5 ISA):
//  A (16x32 f16): row m = lane&15; halves j<8 -> K = sel*8+j ; j>=8 -> K = 16+sel*8+(j-8)
//  B (32x16 f16): col n = lane&15; halves j<8 -> K = sel*8+j ; j>=8 -> K = 16+sel*8+(j-8)
//  C (16x16 f32): row = sel*8 + r (r = VGPR index), col = lane&15
// ---------------------------------------------------------------------------
static __device__ __forceinline__ v16h lda(const _Float16* tile, int stride,
                                           int m, int sel, int koff) {
  const _Float16* row = tile + m * stride + koff;
  v8h lo = *(const v8h*)(row + sel * 8);
  v8h hi = *(const v8h*)(row + 16 + sel * 8);
  v16h a;
#pragma unroll
  for (int i = 0; i < 8; ++i) { a[i] = lo[i]; a[i + 8] = hi[i]; }
  return a;
}

static __device__ __forceinline__ v16h ldb(const _Float16* tileT, int stride,
                                           int n, int sel, int koff) {
  const _Float16* col = tileT + n * stride + koff;
  v8h lo = *(const v8h*)(col + sel * 8);
  v8h hi = *(const v8h*)(col + 16 + sel * 8);
  v16h b;
#pragma unroll
  for (int i = 0; i < 8; ++i) { b[i] = lo[i]; b[i + 8] = hi[i]; }
  return b;
}

static __device__ __forceinline__ v8f wmma_f16(v16h a, v16h b, v8f c) {
  return __builtin_amdgcn_wmma_f32_16x16x32_f16(false, a, false, b,
                                                (short)0, c, false, false);
}

// ---------------------------------------------------------------------------
// TDM: issue a 2D tensor_load_to_lds (CDNA5 ISA).
//  dataSizeLog2: 0=1B 1=2B 2=4B 3=8B (elements)
//  tile: tileX elems/row, tileY rows; global row stride = strideX elems
//  tensor dims set to the exact tile (no OOB), trailing dims 0 (unused).
// ---------------------------------------------------------------------------
static __device__ __forceinline__ void tdm_load_2d(unsigned ldsOff,
                                                   unsigned long long gaddr,
                                                   int dataSizeLog2,
                                                   int tileX, int tileY,
                                                   long strideX) {
  u32x4 g0;
  g0[0] = 1u;                                         // count=1 (valid), user mode
  g0[1] = ldsOff;                                     // lds_addr [63:32]
  g0[2] = (unsigned)(gaddr & 0xffffffffull);          // global_addr lo
  g0[3] = (unsigned)((gaddr >> 32) & 0x01ffffffull)   // global_addr [56:32]
          | 0x80000000u;                              // type=2 ("image") [127:126]
  i32x8 g1;
  g1[0] = dataSizeLog2 << 16;                         // wg_mask=0, data_size
  g1[1] = (tileX & 0xffff) << 16;                     // tensor_dim0[15:0] @ [63:48]
  g1[2] = ((unsigned)tileX >> 16) | ((tileY & 0xffff) << 16); // tdim0 hi | tdim1 lo
  g1[3] = ((unsigned)tileY >> 16) | ((tileX & 0xffff) << 16); // tdim1 hi | tile_dim0
  g1[4] = tileY & 0xffff;                             // tile_dim1 | tile_dim2=0
  g1[5] = (int)(strideX & 0xffffffffll);              // tensor_dim0_stride lo32
  g1[6] = (int)((strideX >> 32) & 0xffffll);          // stride0 hi16 | stride1 lo=0
  g1[7] = 0;                                          // stride1 hi
  i32x4 gz4 = {0, 0, 0, 0};
  i32x8 gz8 = {0, 0, 0, 0, 0, 0, 0, 0};
  __builtin_amdgcn_tensor_load_to_lds(g0, g1, gz4, gz4, gz8, 0);
}

static __device__ __forceinline__ unsigned lds_offset(const void* p) {
  return (unsigned)(unsigned long long)(uintptr_t)p;  // generic->LDS: addr[31:0]
}

// ---------------------------------------------------------------------------
// Kernel 1: qkv = x @ W_attn + b_attn  -> q,k,v f16 in [B,H,T,64] layout
// M=8192, N=2304, K=768.  Block: 256 thr (8 waves), tile 64(M) x 128(N), K-step 32.
// (both tiles need f32->f16 conversion, so staging stays on the VALU path)
// ---------------------------------------------------------------------------
__global__ __launch_bounds__(256) void qkv_gemm_kernel(
    const float* __restrict__ x, const float* __restrict__ W,
    const float* __restrict__ bias,
    _Float16* __restrict__ qb, _Float16* __restrict__ kb,
    _Float16* __restrict__ vb) {
  __shared__ __attribute__((aligned(32))) _Float16 As[64 * 32];    // [m][k]
  __shared__ __attribute__((aligned(32))) _Float16 Bt[128 * 32];   // [n][k]

  const int mbase = blockIdx.x * 64;
  const int nbase = blockIdx.y * 128;
  const int tid = threadIdx.x, lane = tid & 31, wv = tid >> 5;
  const int mi = wv & 3, ni = wv >> 2;
  const int l15 = lane & 15, sel = lane >> 4;

  const v8f zero = {0.f, 0.f, 0.f, 0.f, 0.f, 0.f, 0.f, 0.f};
  v8f acc[4] = {zero, zero, zero, zero};

  for (int kk = 0; kk < DMODEL; kk += 32) {
    __syncthreads();
    for (int i = tid; i < 512; i += 256) {
      int row = i >> 3, c4 = i & 7;
      float4 f = *(const float4*)(x + (size_t)(mbase + row) * DMODEL + kk + c4 * 4);
      _Float16* d = &As[row * 32 + c4 * 4];
      d[0] = (_Float16)f.x; d[1] = (_Float16)f.y;
      d[2] = (_Float16)f.z; d[3] = (_Float16)f.w;
    }
    for (int i = tid; i < 1024; i += 256) {
      int kr = i >> 5, c4 = i & 31;
      float4 f = *(const float4*)(W + (size_t)(kk + kr) * 2304 + nbase + c4 * 4);
      Bt[(c4 * 4 + 0) * 32 + kr] = (_Float16)f.x;
      Bt[(c4 * 4 + 1) * 32 + kr] = (_Float16)f.y;
      Bt[(c4 * 4 + 2) * 32 + kr] = (_Float16)f.z;
      Bt[(c4 * 4 + 3) * 32 + kr] = (_Float16)f.w;
    }
    if (kk + 32 < DMODEL)  // stream next weight tile into L2 (global_prefetch_b8)
      __builtin_prefetch(W + (size_t)(kk + 32 + (tid >> 3)) * 2304 + nbase + (tid & 7) * 16, 0, 1);
    __syncthreads();

    v16h a = lda(&As[mi * 16 * 32], 32, l15, sel, 0);
#pragma unroll
    for (int t = 0; t < 4; ++t) {
      v16h b = ldb(&Bt[(ni * 64 + t * 16) * 32], 32, l15, sel, 0);
      acc[t] = wmma_f16(a, b, acc[t]);
    }
  }

#pragma unroll
  for (int t = 0; t < 4; ++t) {
    int n = nbase + ni * 64 + t * 16 + l15;
    float bv = bias[n];
    int sec = n / DMODEL;
    int nin = n - sec * DMODEL;
    int h = nin >> 6, d = nin & 63;
    _Float16* dst = (sec == 0) ? qb : (sec == 1) ? kb : vb;
#pragma unroll
    for (int r = 0; r < 8; ++r) {
      int m = mbase + mi * 16 + sel * 8 + r;
      int bidx = m >> 12;
      int tpos = m & (SEQT - 1);
      dst[(((size_t)bidx * NHEAD + h) * SEQT + tpos) * HDIM + d] =
          (_Float16)(acc[t][r] + bv);
    }
  }
}

// ---------------------------------------------------------------------------
// Kernel 2: causal flash attention.  Block = 128 thr (4 waves) owns one
// (b, h, 64-row Q tile); wave owns 16 Q rows.  Key blocks of 32 keys.
// K tile staged by the Tensor Data Mover; V transposed manually (overlapped).
// ---------------------------------------------------------------------------
__global__ __launch_bounds__(128) void flash_attn_kernel(
    const _Float16* __restrict__ qb, const _Float16* __restrict__ kb,
    const _Float16* __restrict__ vb, const int* __restrict__ amask,
    _Float16* __restrict__ att_out) {
  __shared__ __attribute__((aligned(32))) _Float16 Ks[32 * 64];      // [key][d]
  __shared__ __attribute__((aligned(32))) _Float16 Vt[64 * 32];      // [d][key]
  __shared__ __attribute__((aligned(32))) _Float16 Ps[4 * 16 * 32];  // per-wave P

  const int qbase = blockIdx.x * 64;
  const int h = blockIdx.y, b = blockIdx.z;
  const int tid = threadIdx.x, lane = tid & 31, wv = tid >> 5;
  const int l15 = lane & 15, sel = lane >> 4;
  const size_t headBase = ((size_t)(b * NHEAD + h)) * SEQT * HDIM;
  const int qw = qbase + wv * 16;
  const float scale = 0.125f;
  const unsigned ldsKs = lds_offset(&Ks[0]);

  const _Float16* qrow = qb + headBase + (size_t)(qw + l15) * HDIM;
  v16h qa0, qa1;
  {
    v8h lo = *(const v8h*)(qrow + sel * 8);
    v8h hi = *(const v8h*)(qrow + 16 + sel * 8);
#pragma unroll
    for (int i = 0; i < 8; ++i) { qa0[i] = lo[i]; qa0[i + 8] = hi[i]; }
    lo = *(const v8h*)(qrow + 32 + sel * 8);
    hi = *(const v8h*)(qrow + 48 + sel * 8);
#pragma unroll
    for (int i = 0; i < 8; ++i) { qa1[i] = lo[i]; qa1[i + 8] = hi[i]; }
  }

  const v8f zero = {0.f, 0.f, 0.f, 0.f, 0.f, 0.f, 0.f, 0.f};
  float mrow[8], lrow[8];
  v8f o[4] = {zero, zero, zero, zero};
#pragma unroll
  for (int r = 0; r < 8; ++r) { mrow[r] = -3.0e38f; lrow[r] = 0.f; }

  const int nkb = qbase / 32 + 2;

  for (int kbi = 0; kbi < nkb; ++kbi) {
    __syncthreads();
    const int keyBase = kbi * 32;

    // TDM: K tile [32 keys x 64 d] f16 == 32 rows of 16 x 8B elements, contiguous
    if (wv == 0) {
      tdm_load_2d(ldsKs,
                  (unsigned long long)(uintptr_t)(kb + headBase + (size_t)keyBase * HDIM),
                  /*8B*/ 3, /*tileX*/ 16, /*tileY*/ 32, /*strideX*/ 16);
    }
    // overlap: stage V transposed via VALU path
    for (int i = tid; i < 256; i += 128) {
      int row = i >> 3, c8 = i & 7;
      v8h f = *(const v8h*)(vb + headBase + (size_t)(keyBase + row) * HDIM + c8 * 8);
#pragma unroll
      for (int j = 0; j < 8; ++j) Vt[(c8 * 8 + j) * 32 + row] = f[j];
    }
    if (wv == 0) __builtin_amdgcn_s_wait_tensorcnt(0);
    __syncthreads();

    v8f s0 = wmma_f16(qa0, ldb(&Ks[0], 64, l15, sel, 0), zero);
    s0 = wmma_f16(qa1, ldb(&Ks[0], 64, l15, sel, 32), s0);
    v8f s1 = wmma_f16(qa0, ldb(&Ks[16 * 64], 64, l15, sel, 0), zero);
    s1 = wmma_f16(qa1, ldb(&Ks[16 * 64], 64, l15, sel, 32), s1);

    const int kg0 = keyBase + l15, kg1 = kg0 + 16;
    const float pad0 = amask[b * SEQT + kg0] ? 0.f : -1e9f;
    const float pad1 = amask[b * SEQT + kg1] ? 0.f : -1e9f;

    float p0a[8], p1a[8], alpha[8];
#pragma unroll
    for (int r = 0; r < 8; ++r) {
      int qg = qw + sel * 8 + r;
      float v0 = s0[r] * scale + pad0 + ((kg0 > qg) ? -10000.f : 0.f);
      float v1 = s1[r] * scale + pad1 + ((kg1 > qg) ? -10000.f : 0.f);
      float mx = fmaxf(v0, v1);
#pragma unroll
      for (int off = 8; off >= 1; off >>= 1) mx = fmaxf(mx, __shfl_xor(mx, off, 16));
      float mnew = fmaxf(mrow[r], mx);
      float a = __expf(mrow[r] - mnew);
      float e0 = __expf(v0 - mnew);
      float e1 = __expf(v1 - mnew);
      float rs = e0 + e1;
#pragma unroll
      for (int off = 8; off >= 1; off >>= 1) rs += __shfl_xor(rs, off, 16);
      lrow[r] = lrow[r] * a + rs;
      mrow[r] = mnew;
      alpha[r] = a;
      p0a[r] = e0; p1a[r] = e1;
    }

    _Float16* P = &Ps[wv * 16 * 32];
#pragma unroll
    for (int r = 0; r < 8; ++r) {
      int row = sel * 8 + r;
      P[row * 32 + l15] = (_Float16)p0a[r];
      P[row * 32 + 16 + l15] = (_Float16)p1a[r];
#pragma unroll
      for (int t = 0; t < 4; ++t) o[t][r] *= alpha[r];
    }
    __syncthreads();

    v16h pa = lda(P, 32, l15, sel, 0);
#pragma unroll
    for (int t = 0; t < 4; ++t) {
      v16h vbF = ldb(&Vt[(t * 16) * 32], 32, l15, sel, 0);
      o[t] = wmma_f16(pa, vbF, o[t]);
    }
  }

#pragma unroll
  for (int r = 0; r < 8; ++r) {
    float inv = 1.0f / lrow[r];
    int qg = qw + sel * 8 + r;
#pragma unroll
    for (int t = 0; t < 4; ++t) {
      int d = t * 16 + l15;
      att_out[((size_t)(b * SEQT + qg)) * DMODEL + h * HDIM + d] =
          (_Float16)(o[t][r] * inv);
    }
  }
}

// ---------------------------------------------------------------------------
// Kernel 3: out = att @ W_proj + b_proj.  M=8192, N=768, K=768.
// A tile (pure f16 copy, stride 768) staged by the Tensor Data Mover.
// ---------------------------------------------------------------------------
__global__ __launch_bounds__(256) void proj_gemm_kernel(
    const _Float16* __restrict__ att, const float* __restrict__ W,
    const float* __restrict__ bias, float* __restrict__ out) {
  __shared__ __attribute__((aligned(32))) _Float16 As[64 * 32];
  __shared__ __attribute__((aligned(32))) _Float16 Bt[128 * 32];

  const int mbase = blockIdx.x * 64;
  const int nbase = blockIdx.y * 128;
  const int tid = threadIdx.x, lane = tid & 31, wv = tid >> 5;
  const int mi = wv & 3, ni = wv >> 2;
  const int l15 = lane & 15, sel = lane >> 4;
  const unsigned ldsAs = lds_offset(&As[0]);

  const v8f zero = {0.f, 0.f, 0.f, 0.f, 0.f, 0.f, 0.f, 0.f};
  v8f acc[4] = {zero, zero, zero, zero};

  for (int kk = 0; kk < DMODEL; kk += 32) {
    __syncthreads();
    // TDM: A tile = 64 rows x 32 f16 elems out of a 768-stride f16 tensor
    if (wv == 0) {
      tdm_load_2d(ldsAs,
                  (unsigned long long)(uintptr_t)(att + (size_t)mbase * DMODEL + kk),
                  /*2B*/ 1, /*tileX*/ 32, /*tileY*/ 64, /*strideX*/ DMODEL);
    }
    for (int i = tid; i < 1024; i += 256) {
      int kr = i >> 5, c4 = i & 31;
      float4 f = *(const float4*)(W + (size_t)(kk + kr) * DMODEL + nbase + c4 * 4);
      Bt[(c4 * 4 + 0) * 32 + kr] = (_Float16)f.x;
      Bt[(c4 * 4 + 1) * 32 + kr] = (_Float16)f.y;
      Bt[(c4 * 4 + 2) * 32 + kr] = (_Float16)f.z;
      Bt[(c4 * 4 + 3) * 32 + kr] = (_Float16)f.w;
    }
    if (kk + 32 < DMODEL)
      __builtin_prefetch(W + (size_t)(kk + 32 + (tid >> 3)) * DMODEL + nbase + (tid & 7) * 16, 0, 1);
    if (wv == 0) __builtin_amdgcn_s_wait_tensorcnt(0);
    __syncthreads();

    v16h a = lda(&As[mi * 16 * 32], 32, l15, sel, 0);
#pragma unroll
    for (int t = 0; t < 4; ++t) {
      v16h b = ldb(&Bt[(ni * 64 + t * 16) * 32], 32, l15, sel, 0);
      acc[t] = wmma_f16(a, b, acc[t]);
    }
  }

#pragma unroll
  for (int t = 0; t < 4; ++t) {
    int n = nbase + ni * 64 + t * 16 + l15;
    float bv = bias[n];
#pragma unroll
    for (int r = 0; r < 8; ++r) {
      int m = mbase + mi * 16 + sel * 8 + r;
      out[(size_t)m * DMODEL + n] = acc[t][r] + bv;
    }
  }
}

// ---------------------------------------------------------------------------
extern "C" void kernel_launch(void* const* d_in, const int* in_sizes, int n_in,
                              void* d_out, int out_size, void* d_ws, size_t ws_size,
                              hipStream_t stream) {
  (void)in_sizes; (void)n_in; (void)out_size; (void)ws_size;
  const float* x      = (const float*)d_in[0];
  const int*   amask  = (const int*)d_in[1];
  const float* W_attn = (const float*)d_in[2];
  const float* b_attn = (const float*)d_in[3];
  const float* W_proj = (const float*)d_in[4];
  const float* b_proj = (const float*)d_in[5];
  float* out = (float*)d_out;

  char* ws = (char*)d_ws;
  const size_t seg = (size_t)BATCH * NHEAD * SEQT * HDIM * sizeof(_Float16);
  _Float16* qb  = (_Float16*)(ws);
  _Float16* kb  = (_Float16*)(ws + seg);
  _Float16* vb  = (_Float16*)(ws + 2 * seg);
  _Float16* att = (_Float16*)(ws + 3 * seg);

  qkv_gemm_kernel<<<dim3(128, 18), 256, 0, stream>>>(x, W_attn, b_attn, qb, kb, vb);
  flash_attn_kernel<<<dim3(SEQT / 64, NHEAD, BATCH), 128, 0, stream>>>(qb, kb, vb, amask, att);
  proj_gemm_kernel<<<dim3(128, 6), 256, 0, stream>>>(att, W_proj, b_proj, out);
}
